// HeteroPULLModel_24489903522445
// MI455X (gfx1250) — matrix operand order repaired and checked by reading, not compiled, not run on hardware
//
#include <hip/hip_runtime.h>
#include <hip/hip_bf16.h>
#include <math.h>

// ---------------- fixed model metadata (matches reference) ----------------
#define HID   128
#define HEADS 4
#define DHH   32
#define OUTD  64
#define NC    15000
#define ND    3000
#define NG    40000
#define NTOT  58000            // NC+ND+NG
#define KADD  8000
#define SORTN 8192
#define MBLK  4                // M-tiles per wave (B-tile reuse + WMMA pipelining)

typedef __attribute__((ext_vector_type(16))) _Float16 v16h;
typedef __attribute__((ext_vector_type(8)))  float    v8f;

// ---------------- helpers ----------------
__device__ __forceinline__ unsigned encf(float f) {
  unsigned u = __float_as_uint(f);
  return (u & 0x80000000u) ? ~u : (u | 0x80000000u);
}
__device__ __forceinline__ float decf(unsigned u) {
  return (u & 0x80000000u) ? __uint_as_float(u & 0x7FFFFFFFu)
                           : __uint_as_float(~u);
}
__device__ __forceinline__ float gelu_f(float x) {
  return 0.5f * x * (1.0f + erff(x * 0.70710678118654752f));
}

// pack 16 floats (4x float4) into the f16 vector, optional fused GELU
template<bool GELU>
__device__ __forceinline__ v16h pack16(float4 a0, float4 a1, float4 a2, float4 a3) {
  float t[16] = {a0.x, a0.y, a0.z, a0.w, a1.x, a1.y, a1.z, a1.w,
                 a2.x, a2.y, a2.z, a2.w, a3.x, a3.y, a3.z, a3.w};
  v16h a;
#pragma unroll
  for (int j = 0; j < 16; ++j) {
    float f = t[j];
    if (GELU) f = gelu_f(f);
    a[j] = (_Float16)f;
  }
  return a;
}

// A-matrix 16x32 f16 tile (ISA 7.12.2): lane holds row M=lane&15; halves are
// two contiguous 8-float runs at +laneoff and +laneoff+16 (laneoff = 8*(lane>=16)).
// Loads as 4x global_load_b128 (16B aligned for all operands in this model).
template<bool GELU>
__device__ __forceinline__ v16h load_a_vec(const float* A, int lda, int M, int row0, int k0) {
  int lane = threadIdx.x & 31;
  int m = row0 + (lane & 15);
  if (m >= M) m = M - 1;                       // clamp; masked at store
  const float4* p = (const float4*)(A + (long)m * lda + k0 + ((lane & 16) >> 1));
  return pack16<GELU>(p[0], p[1], p[4], p[5]);
}

// B row-major [K,N]; B-matrix 32x16 f16: lane = column n0+(lane&15),
// K-half by lane group. Strided (ld = N) -> scalar loads, amortized via MBLK.
__device__ __forceinline__ v16h load_b(const float* B, int ldb, int N, int k0, int n0) {
  int lane = threadIdx.x & 31;
  int n = n0 + (lane & 15);
  if (n >= N) n = N - 1;
  int kb = k0 + ((lane >> 4) << 3);
  const float* p = B + (long)kb * ldb + n;
  v16h b;
#pragma unroll
  for (int j = 0; j < 16; ++j) b[j] = (_Float16)p[(long)(j < 8 ? j : j + 8) * ldb];
  return b;
}

// B transposed: Bt row-major [N,K]; B[k][n] = Bt[n][k] -> two 8-float runs at k0+8h and k0+8h+16.
__device__ __forceinline__ v16h load_bt_vec(const float* Bt, int ldbt, int N, int k0, int n0) {
  int lane = threadIdx.x & 31;
  int n = n0 + (lane & 15);
  if (n >= N) n = N - 1;
  int kb = k0 + ((lane >> 4) << 3);
  const float4* p = (const float4*)(Bt + (long)n * ldbt + kb);
  return pack16<false>(p[0], p[1], p[4], p[5]);
}

// ---------------- GEMM kernels: MBLK 16x16 tiles per wave32 ----------------
__global__ void k_gemm_bias(const float* __restrict__ A, const float* __restrict__ B,
                            const float* __restrict__ bias, float* __restrict__ C,
                            int M, int N, int K) {
  int tilesN = (N + 15) >> 4, tilesM = (M + 15) >> 4;
  int mg = (tilesM + MBLK - 1) / MBLK;
  int tile = blockIdx.x * (blockDim.x >> 5) + (threadIdx.x >> 5);
  if (tile >= mg * tilesN) return;             // wave-uniform
  int tm0 = (tile / tilesN) * MBLK, tn = tile % tilesN;
  int nm = tilesM - tm0; if (nm > MBLK) nm = MBLK;   // uniform
  v8f acc[MBLK] = {};
  for (int k0 = 0; k0 < K; k0 += 32) {
    v16h b = load_b(B, N, N, k0, tn * 16);
#pragma unroll
    for (int i = 0; i < MBLK; ++i) {
      if (i < nm) {
        v16h a = load_a_vec<false>(A, K, M, (tm0 + i) * 16, k0);
        acc[i] = __builtin_amdgcn_wmma_f32_16x16x32_f16(false, a, false, b, (short)0,
                                                        acc[i], false, false);
      }
    }
  }
  int lane = threadIdx.x & 31;
  int n = tn * 16 + (lane & 15);
  float bv = (n < N) ? bias[n] : 0.0f;
#pragma unroll
  for (int i = 0; i < MBLK; ++i) {
    if (i >= nm) break;
    int mb = (tm0 + i) * 16 + ((lane & 16) >> 1);
    if (n < N) {
      if (mb + 8 <= M) {                        // fast path: unguarded stores
#pragma unroll
        for (int r = 0; r < 8; ++r) C[(long)(mb + r) * N + n] = acc[i][r] + bv;
      } else {
#pragma unroll
        for (int r = 0; r < 8; ++r)
          if (mb + r < M) C[(long)(mb + r) * N + n] = acc[i][r] + bv;
      }
    }
  }
}

__global__ void k_gemm_gelu_skip(const float* __restrict__ A, const float* __restrict__ B,
                                 const float* __restrict__ bias,
                                 const float* __restrict__ xold,
                                 const float* __restrict__ skipPtr,
                                 float* __restrict__ C, int M, int N, int K) {
  int tilesN = (N + 15) >> 4, tilesM = (M + 15) >> 4;
  int mg = (tilesM + MBLK - 1) / MBLK;
  int tile = blockIdx.x * (blockDim.x >> 5) + (threadIdx.x >> 5);
  if (tile >= mg * tilesN) return;
  int tm0 = (tile / tilesN) * MBLK, tn = tile % tilesN;
  int nm = tilesM - tm0; if (nm > MBLK) nm = MBLK;
  v8f acc[MBLK] = {};
  for (int k0 = 0; k0 < K; k0 += 32) {
    v16h b = load_b(B, N, N, k0, tn * 16);
#pragma unroll
    for (int i = 0; i < MBLK; ++i) {
      if (i < nm) {
        v16h a = load_a_vec<true>(A, K, M, (tm0 + i) * 16, k0);   // fused GELU
        acc[i] = __builtin_amdgcn_wmma_f32_16x16x32_f16(false, a, false, b, (short)0,
                                                        acc[i], false, false);
      }
    }
  }
  float s = 1.0f / (1.0f + __expf(-skipPtr[0]));
  int lane = threadIdx.x & 31;
  int n = tn * 16 + (lane & 15);
  float bv = (n < N) ? bias[n] : 0.0f;
#pragma unroll
  for (int i = 0; i < MBLK; ++i) {
    if (i >= nm) break;
    int mb = (tm0 + i) * 16 + ((lane & 16) >> 1);
    if (n < N) {
      if (mb + 8 <= M) {
#pragma unroll
        for (int r = 0; r < 8; ++r) {
          long o = (long)(mb + r) * N + n;
          C[o] = s * (acc[i][r] + bv) + (1.0f - s) * xold[o];
        }
      } else {
#pragma unroll
        for (int r = 0; r < 8; ++r) {
          if (mb + r < M) {
            long o = (long)(mb + r) * N + n;
            C[o] = s * (acc[i][r] + bv) + (1.0f - s) * xold[o];
          }
        }
      }
    }
  }
}

// decoder: C[M,N] = A[M,K] @ Bt[N,K]^T  (raw logits)
__global__ void k_gemm_nt(const float* __restrict__ A, const float* __restrict__ Bt,
                          float* __restrict__ C, int M, int N, int K) {
  int tilesN = (N + 15) >> 4, tilesM = (M + 15) >> 4;
  int mg = (tilesM + MBLK - 1) / MBLK;
  int tile = blockIdx.x * (blockDim.x >> 5) + (threadIdx.x >> 5);
  if (tile >= mg * tilesN) return;
  int tm0 = (tile / tilesN) * MBLK, tn = tile % tilesN;
  int nm = tilesM - tm0; if (nm > MBLK) nm = MBLK;
  v8f acc[MBLK] = {};
  for (int k0 = 0; k0 < K; k0 += 32) {
    v16h b = load_bt_vec(Bt, K, N, k0, tn * 16);
#pragma unroll
    for (int i = 0; i < MBLK; ++i) {
      if (i < nm) {
        v16h a = load_a_vec<false>(A, K, M, (tm0 + i) * 16, k0);
        acc[i] = __builtin_amdgcn_wmma_f32_16x16x32_f16(false, a, false, b, (short)0,
                                                        acc[i], false, false);
      }
    }
  }
  int lane = threadIdx.x & 31;
  int n = tn * 16 + (lane & 15);
#pragma unroll
  for (int i = 0; i < MBLK; ++i) {
    if (i >= nm) break;
    int mb = (tm0 + i) * 16 + ((lane & 16) >> 1);
    if (n < N) {
      if (mb + 8 <= M) {
#pragma unroll
        for (int r = 0; r < 8; ++r) C[(long)(mb + r) * N + n] = acc[i][r];
      } else {
#pragma unroll
        for (int r = 0; r < 8; ++r)
          if (mb + r < M) C[(long)(mb + r) * N + n] = acc[i][r];
      }
    }
  }
}

// ---------------- edge kernels ----------------
// alpha pass: k_e = gather(k[src]) @ a_rel[h] (WMMA), alpha = <q[dst], k_e> * p * scale
__global__ void k_edge_alpha(const float* __restrict__ kqv_src, const float* __restrict__ kqv_dst,
                             const int* __restrict__ src, const int* __restrict__ dst, int E,
                             const float* __restrict__ a_rel,   // [H][32][32]
                             const float* __restrict__ p_rel,   // [H]
                             float* __restrict__ alpha,         // [E][H]
                             unsigned* __restrict__ mmax)       // [n_dst][H] encoded max
{
  int e0 = blockIdx.x * 16;
  int lane = threadIdx.x & 31;
  const float scale = 0.17677669529663687f;    // 1/sqrt(32)
  int em = e0 + (lane & 15); if (em >= E) em = E - 1;
  long srow = (long)src[em] * 384;
  for (int h = 0; h < HEADS; ++h) {
    const float4* p4 = (const float4*)(kqv_src + srow + h * DHH + ((lane & 16) >> 1));
    v16h a = pack16<false>(p4[0], p4[1], p4[4], p4[5]);
    float part[8];
#pragma unroll
    for (int r = 0; r < 8; ++r) part[r] = 0.0f;
    for (int nt = 0; nt < 2; ++nt) {
      v16h b = load_b(a_rel + h * 1024, 32, 32, 0, nt * 16);
      v8f d = {};
      d = __builtin_amdgcn_wmma_f32_16x16x32_f16(false, a, false, b, (short)0, d, false, false);
      int n = nt * 16 + (lane & 15);
#pragma unroll
      for (int r = 0; r < 8; ++r) {
        int ei = e0 + r + ((lane & 16) >> 1); if (ei >= E) ei = E - 1;
        float qv = kqv_dst[(long)dst[ei] * 384 + HID + h * DHH + n];
        part[r] += d[r] * qv;
      }
    }
#pragma unroll
    for (int x = 1; x < 16; x <<= 1) {
#pragma unroll
      for (int r = 0; r < 8; ++r) part[r] += __shfl_xor(part[r], x, 32);
    }
    if ((lane & 15) == 0) {
      float pr = p_rel[h];
#pragma unroll
      for (int r = 0; r < 8; ++r) {
        int ei = e0 + r + ((lane & 16) >> 1);
        if (ei < E) {
          float al = part[r] * pr * scale;
          alpha[(long)ei * HEADS + h] = al;
          atomicMax(&mmax[(long)dst[ei] * HEADS + h], encf(al));
        }
      }
    }
  }
}

__global__ void k_exp_den(float* __restrict__ alpha, const int* __restrict__ dst,
                          const unsigned* __restrict__ mmax, float* __restrict__ den, int E) {
  long i = (long)blockIdx.x * blockDim.x + threadIdx.x;
  if (i >= (long)E * HEADS) return;
  int ei = (int)(i >> 2), h = (int)(i & 3);
  int d = dst[ei];
  float ex = __expf(alpha[i] - decf(mmax[(long)d * HEADS + h]));
  alpha[i] = ex;
  atomicAdd(&den[(long)d * HEADS + h], ex);
}

// agg pass: v_e = gather(v[src]) @ m_rel[h] (WMMA); agg[dst] += (ex/den) * v_e
__global__ void k_edge_agg(const float* __restrict__ kqv_src,
                           const int* __restrict__ src, const int* __restrict__ dst, int E,
                           const float* __restrict__ m_rel,    // [H][32][32]
                           const float* __restrict__ ex,       // [E][H]
                           const float* __restrict__ den,      // [n_dst][H]
                           float* __restrict__ agg)            // [n_dst][HID]
{
  int e0 = blockIdx.x * 16;
  int lane = threadIdx.x & 31;
  int em = e0 + (lane & 15); if (em >= E) em = E - 1;
  long srow = (long)src[em] * 384;
  for (int h = 0; h < HEADS; ++h) {
    const float4* p4 = (const float4*)(kqv_src + srow + 2 * HID + h * DHH + ((lane & 16) >> 1));
    v16h a = pack16<false>(p4[0], p4[1], p4[4], p4[5]);
    for (int nt = 0; nt < 2; ++nt) {
      v16h b = load_b(m_rel + h * 1024, 32, 32, 0, nt * 16);
      v8f d = {};
      d = __builtin_amdgcn_wmma_f32_16x16x32_f16(false, a, false, b, (short)0, d, false, false);
      int n = nt * 16 + (lane & 15);
#pragma unroll
      for (int r = 0; r < 8; ++r) {
        int ei = e0 + r + ((lane & 16) >> 1);
        if (ei < E) {
          int dn = dst[ei];
          float w = ex[(long)ei * HEADS + h] / (den[(long)dn * HEADS + h] + 1e-16f);
          atomicAdd(&agg[(long)dn * HID + h * DHH + n], w * d[r]);
        }
      }
    }
  }
}

// ---------------- decode / top-k ----------------
__global__ void k_mask_train(float* __restrict__ prob, const int* __restrict__ ei, int E) {
  int i = blockIdx.x * blockDim.x + threadIdx.x;
  if (i < E) prob[(long)ei[i] * ND + ei[E + i]] = -__builtin_huge_valf();
}

__global__ void k_hist(const float* __restrict__ lg, long n, unsigned* __restrict__ hist) {
  long i = (long)blockIdx.x * blockDim.x + threadIdx.x;
  if (i < n) atomicAdd(&hist[encf(lg[i]) >> 16], 1u);
}

__global__ void k_scan1(const unsigned* __restrict__ hist, unsigned* __restrict__ scr) {
  if (threadIdx.x == 0 && blockIdx.x == 0) {
    unsigned cum = 0, above = 0; int T = 0;
    for (int b = 65535; b >= 0; --b) {
      unsigned c = hist[b];
      if (cum + c >= (unsigned)KADD) { T = b; above = cum; break; }
      cum += c;
    }
    scr[0] = (unsigned)T; scr[1] = above; scr[2] = (unsigned)KADD - above;
  }
}

__global__ void k_hist2(const float* __restrict__ lg, long n,
                        const unsigned* __restrict__ scr, unsigned* __restrict__ hist2) {
  long i = (long)blockIdx.x * blockDim.x + threadIdx.x;
  if (i >= n) return;
  unsigned u = encf(lg[i]);
  if ((u >> 16) == scr[0]) atomicAdd(&hist2[u & 0xFFFFu], 1u);
}

__global__ void k_scan2(const unsigned* __restrict__ hist2, unsigned* __restrict__ scr) {
  if (threadIdx.x == 0 && blockIdx.x == 0) {
    unsigned need = scr[2], cum = 0, above = 0; int T2 = 0;
    for (int b = 65535; b >= 0; --b) {
      unsigned c = hist2[b];
      if (cum + c >= need) { T2 = b; above = cum; break; }
      cum += c;
    }
    scr[3] = (scr[0] << 16) | (unsigned)T2;   // exact 32-bit threshold key
    scr[4] = scr[1] + above;                  // # strictly greater
    scr[5] = (unsigned)KADD - scr[4];         // # equals to take
    scr[6] = 0u; scr[7] = 0u;                 // collection counters
  }
}

__global__ void k_collect(const float* __restrict__ lg, long n,
                          unsigned* __restrict__ scr, uint2* __restrict__ sel) {
  long i = (long)blockIdx.x * blockDim.x + threadIdx.x;
  if (i >= n) return;
  unsigned u = encf(lg[i]);
  unsigned key = scr[3];
  if (u > key) {
    unsigned p = atomicAdd(&scr[6], 1u);
    if (p < (unsigned)KADD) sel[p] = make_uint2(u, (unsigned)i);
  } else if (u == key) {
    unsigned t = atomicAdd(&scr[7], 1u);
    if (t < scr[5]) sel[scr[4] + t] = make_uint2(u, (unsigned)i);
  }
}

// single-block bitonic sort of 8192 (key,idx) pairs; outputs descending top-KADD
__global__ __launch_bounds__(1024) void k_sort_out(const uint2* __restrict__ sel,
                                                   float* __restrict__ out) {
  __shared__ unsigned sk[SORTN];
  __shared__ unsigned si[SORTN];
  for (int i = threadIdx.x; i < SORTN; i += blockDim.x) {
    if (i < KADD) { sk[i] = sel[i].x; si[i] = sel[i].y; }
    else          { sk[i] = 0u;       si[i] = 0u; }
  }
  __syncthreads();
  for (unsigned k = 2; k <= SORTN; k <<= 1) {
    for (unsigned j = k >> 1; j > 0; j >>= 1) {
      for (unsigned i = threadIdx.x; i < SORTN; i += blockDim.x) {
        unsigned l = i ^ j;
        if (l > i) {
          bool up = ((i & k) == 0);
          if ((sk[i] > sk[l]) == up) {
            unsigned tk = sk[i]; sk[i] = sk[l]; sk[l] = tk;
            unsigned ti = si[i]; si[i] = si[l]; si[l] = ti;
          }
        }
      }
      __syncthreads();
    }
  }
  for (int i = threadIdx.x; i < KADD; i += blockDim.x) {
    unsigned u  = sk[SORTN - 1 - i];
    unsigned id = si[SORTN - 1 - i];
    out[i]            = (float)(id / ND);                       // row
    out[KADD + i]     = (float)(id % ND);                       // col
    out[2 * KADD + i] = 1.0f / (1.0f + __expf(-decf(u)));       // prob
  }
}

// ---------------- host ----------------
static inline int cdiv(long a, long b) { return (int)((a + b - 1) / b); }

extern "C" void kernel_launch(void* const* d_in, const int* in_sizes, int n_in,
                              void* d_out, int out_size, void* d_ws, size_t ws_size,
                              hipStream_t stream) {
  (void)in_sizes; (void)n_in; (void)out_size; (void)ws_size;
  const float* emb[3] = {(const float*)d_in[0], (const float*)d_in[1], (const float*)d_in[2]};
  const float* W_kqv = (const float*)d_in[3];
  const float* b_kqv = (const float*)d_in[4];
  const float* W_out = (const float*)d_in[5];
  const float* b_out = (const float*)d_in[6];
  const float* skip  = (const float*)d_in[7];
  const float* a_rel = (const float*)d_in[8];
  const float* m_rel = (const float*)d_in[9];
  const float* p_rel = (const float*)d_in[10];
  const float* W_fin = (const float*)d_in[11];
  const float* b_fin = (const float*)d_in[12];
  const int* ei[5] = {(const int*)d_in[13], (const int*)d_in[14], (const int*)d_in[15],
                      (const int*)d_in[16], (const int*)d_in[17]};
  const int* train_ei = (const int*)d_in[18];

  const int NODE_N[3] = {NC, ND, NG};
  const long O[3] = {0, NC, NC + ND};
  const int ECNT[5] = {80000, 80000, 250000, 250000, 400000};
  const int EST[5] = {0, 1, 0, 2, 2};
  const int EDT[5] = {1, 0, 2, 0, 2};
  long EOFF[5]; long etot = 0;
  for (int e = 0; e < 5; ++e) { EOFF[e] = etot; etot += ECNT[e]; }

  // ---- workspace carve (≈ 275 MB; PROB overlays KQV/AGG region) ----
  char* w = (char*)d_ws;
  auto carve = [&](size_t bytes) { char* p = w; w += (bytes + 255) & ~(size_t)255; return p; };
  float*    X0    = (float*)carve((size_t)NTOT * HID * 4);
  float*    X1    = (float*)carve((size_t)NTOT * HID * 4);
  float*    Z     = (float*)carve((size_t)NTOT * OUTD * 4);
  unsigned* MMAX  = (unsigned*)carve((size_t)NTOT * HEADS * 4);
  float*    DEN   = (float*)carve((size_t)NTOT * HEADS * 4);
  float*    ALPH  = (float*)carve((size_t)etot * HEADS * 4);
  unsigned* SCR   = (unsigned*)carve(256 * 4);
  unsigned* HIST  = (unsigned*)carve(65536 * 4);
  unsigned* HIST2 = (unsigned*)carve(65536 * 4);
  uint2*    SEL   = (uint2*)carve((size_t)SORTN * 8);
  char* overlay = w;
  float* KQV  = (float*)overlay;
  float* AGG  = (float*)(overlay + (size_t)NTOT * 384 * 4);
  float* PROB = (float*)overlay;                 // reuses KQV+AGG after layers

  // ---- init node features from embeddings ----
  for (int t = 0; t < 3; ++t)
    hipMemcpyAsync(X0 + O[t] * HID, emb[t], (size_t)NODE_N[t] * HID * 4,
                   hipMemcpyDeviceToDevice, stream);

  float* Xc = X0; float* Xn = X1;
  for (int l = 0; l < 2; ++l) {
    // kqv GEMMs: [n_t,128] @ [128,384]
    for (int t = 0; t < 3; ++t) {
      int M = NODE_N[t];
      long tiles = (long)cdiv(cdiv(M, 16), MBLK) * (384 / 16);
      k_gemm_bias<<<cdiv(tiles, 4), 128, 0, stream>>>(
          Xc + O[t] * HID, W_kqv + (size_t)(l * 3 + t) * HID * 384,
          b_kqv + (size_t)(l * 3 + t) * 384, KQV + O[t] * 384, M, 384, HID);
    }
    hipMemsetAsync(MMAX, 0, (size_t)NTOT * HEADS * 4, stream);   // 0 < encf(-inf): lowest key, max floor
    hipMemsetAsync(DEN,  0, (size_t)NTOT * HEADS * 4, stream);
    hipMemsetAsync(AGG,  0, (size_t)NTOT * HID * 4, stream);
    for (int e = 0; e < 5; ++e) {
      int E = ECNT[e]; const int* src = ei[e]; const int* dst = ei[e] + E;
      k_edge_alpha<<<cdiv(E, 16), 32, 0, stream>>>(
          KQV + O[EST[e]] * 384, KQV + O[EDT[e]] * 384, src, dst, E,
          a_rel + (size_t)(l * 5 + e) * HEADS * DHH * DHH,
          p_rel + (size_t)(l * 5 + e) * HEADS,
          ALPH + EOFF[e] * HEADS, MMAX + O[EDT[e]] * HEADS);
    }
    for (int e = 0; e < 5; ++e) {
      int E = ECNT[e]; const int* dst = ei[e] + E;
      k_exp_den<<<cdiv((long)E * HEADS, 256), 256, 0, stream>>>(
          ALPH + EOFF[e] * HEADS, dst, MMAX + O[EDT[e]] * HEADS,
          DEN + O[EDT[e]] * HEADS, E);
    }
    for (int e = 0; e < 5; ++e) {
      int E = ECNT[e]; const int* src = ei[e]; const int* dst = ei[e] + E;
      k_edge_agg<<<cdiv(E, 16), 32, 0, stream>>>(
          KQV + O[EST[e]] * 384, src, dst, E,
          m_rel + (size_t)(l * 5 + e) * HEADS * DHH * DHH,
          ALPH + EOFF[e] * HEADS, DEN + O[EDT[e]] * HEADS, AGG + O[EDT[e]] * HID);
    }
    // out-projection with fused GELU + skip blend
    for (int t = 0; t < 3; ++t) {
      int M = NODE_N[t];
      long tiles = (long)cdiv(cdiv(M, 16), MBLK) * (HID / 16);
      k_gemm_gelu_skip<<<cdiv(tiles, 4), 128, 0, stream>>>(
          AGG + O[t] * HID, W_out + (size_t)(l * 3 + t) * HID * HID,
          b_out + (size_t)(l * 3 + t) * HID, Xc + O[t] * HID,
          skip + (l * 3 + t), Xn + O[t] * HID, M, HID, HID);
    }
    float* tmp = Xc; Xc = Xn; Xn = tmp;
  }

  // final per-type linear: [n_t,128] @ [128,64]
  for (int t = 0; t < 3; ++t) {
    int M = NODE_N[t];
    long tiles = (long)cdiv(cdiv(M, 16), MBLK) * (OUTD / 16);
    k_gemm_bias<<<cdiv(tiles, 4), 128, 0, stream>>>(
        Xc + O[t] * HID, W_fin + (size_t)t * HID * OUTD,
        b_fin + (size_t)t * OUTD, Z + O[t] * OUTD, M, OUTD, HID);
  }

  // decoder logits: z_C @ z_D^T  -> PROB [NC, ND]
  {
    long tiles = (long)cdiv(cdiv(NC, 16), MBLK) * cdiv(ND, 16);
    k_gemm_nt<<<cdiv(tiles, 4), 128, 0, stream>>>(Z, Z + O[1] * OUTD, PROB, NC, ND, OUTD);
  }
  k_mask_train<<<cdiv(80000, 256), 256, 0, stream>>>(PROB, train_ei, 80000);

  // global top-KADD via two-level histogram threshold
  long ntotal = (long)NC * ND;
  hipMemsetAsync(HIST, 0, 65536 * 4, stream);
  k_hist<<<cdiv(ntotal, 256), 256, 0, stream>>>(PROB, ntotal, HIST);
  k_scan1<<<1, 1, 0, stream>>>(HIST, SCR);
  hipMemsetAsync(HIST2, 0, 65536 * 4, stream);
  k_hist2<<<cdiv(ntotal, 256), 256, 0, stream>>>(PROB, ntotal, SCR, HIST2);
  k_scan2<<<1, 1, 0, stream>>>(HIST2, SCR);
  k_collect<<<cdiv(ntotal, 256), 256, 0, stream>>>(PROB, ntotal, SCR, SEL);
  k_sort_out<<<1, 1024, 0, stream>>>(SEL, (float*)d_out);
}
